// _HeteroGNNModel_89026082111551
// MI455X (gfx1250) — matrix operand order repaired and checked by reading, not compiled, not run on hardware
//
#include <hip/hip_runtime.h>
#include <hip/hip_bf16.h>

typedef __attribute__((ext_vector_type(16))) _Float16 v16h;
typedef __attribute__((ext_vector_type(8)))  _Float16 v8h;
typedef __attribute__((ext_vector_type(8)))  float    v8f;

#define NUSER 50000
#define NITEM 50000
#define NEDGE 800000
#define HDIM  128

// ---------------------------------------------------------------------------
// A-fragment loader: 16x32 f16 A tile fragment per ISA layout.
// lane 0-15: M=lane, e0..7 -> K=k0..k0+7, e8..15 -> K=k0+16..k0+23
// lane 16-31: same M, K offset +8.  Two 16B contiguous LDS loads per lane.
// ---------------------------------------------------------------------------
__device__ __forceinline__ v16h load_a_frag(const _Float16* base, int lane, int k0, int stride) {
    int m   = lane & 15;
    int khi = (lane >> 4) << 3;
    const _Float16* p = base + m * stride + k0 + khi;
    v8h lo = *(const v8h*)(p);
    v8h hi = *(const v8h*)(p + 16);
    v16h a;
#pragma unroll
    for (int i = 0; i < 8; ++i) { a[i] = lo[i]; a[i + 8] = hi[i]; }
    return a;
}

// ---------------------------------------------------------------------------
// Pre-shuffle a row-major [K,N] f32 weight into WMMA B-fragment order (f16):
// tile (kt,nt) -> 512 halfs; lane l element e = W[kt*32 + (e>=8?16:0) + (l>=16?8:0) + (e&7)][nt*16 + (l&15)]
// Each lane's 16 halfs are contiguous -> B fragment = one 32B load at compute time.
// ---------------------------------------------------------------------------
__global__ void shuffle_weight(const float* __restrict__ W, _Float16* __restrict__ out,
                               int K, int N) {
    int gid = blockIdx.x * blockDim.x + threadIdx.x;
    if (gid >= K * N) return;
    int ntiles = N >> 4;
    int t    = gid >> 9;
    int rif  = gid & 511;
    int lane = rif >> 4;
    int e    = rif & 15;
    int kt = t / ntiles, nt = t - kt * ntiles;
    int k = (kt << 5) + ((e >> 3) << 4) + ((lane >> 4) << 3) + (e & 7);
    int n = (nt << 4) + (lane & 15);
    out[gid] = (_Float16)W[(size_t)k * N + n];
}

__global__ void zero_f32(float* __restrict__ p, size_t n) {
    size_t i  = (size_t)blockIdx.x * blockDim.x + threadIdx.x;
    size_t st = (size_t)gridDim.x * blockDim.x;
    for (; i < n; i += st) p[i] = 0.f;
}

// ---------------------------------------------------------------------------
// Edge scatter: one wave per edge, float4 per lane (128 floats/edge).
// agg[dst] += x[src]; cnt[dst] += 1
// ---------------------------------------------------------------------------
__global__ void scatter_sum(const int* __restrict__ src, const int* __restrict__ dst,
                            const float* __restrict__ xs, float* __restrict__ agg,
                            float* __restrict__ cnt, int ne) {
    int e    = (int)((blockIdx.x * (size_t)blockDim.x + threadIdx.x) >> 5);
    int lane = threadIdx.x & 31;
    if (e >= ne) return;
    int s = src[e], d = dst[e];
    const float4 v = ((const float4*)(xs + (size_t)s * HDIM))[lane];
    float* ap = agg + (size_t)d * HDIM + lane * 4;
    atomicAdd(ap + 0, v.x);
    atomicAdd(ap + 1, v.y);
    atomicAdd(ap + 2, v.z);
    atomicAdd(ap + 3, v.w);
    if (lane == 0) atomicAdd(cnt + d, 1.f);
}

// ---------------------------------------------------------------------------
// Generic WMMA GEMM: Y[M,N] = relu(X[M,K] @ W + bias). N%16==0, K%32==0.
// Block = 32*(N/16) threads, one 16-row tile per block, wave w owns N-tile w.
// ---------------------------------------------------------------------------
__global__ void gemm_bias_relu(const float* __restrict__ X, const _Float16* __restrict__ Wf,
                               const float* __restrict__ bias, float* __restrict__ Y,
                               int M, int K, int N) {
    extern __shared__ _Float16 sA[];       // [16][K]
    const int tile = blockIdx.x, tid = threadIdx.x;
    const int lane = tid & 31, wid = tid >> 5;
    const int r0 = tile << 4;
    for (int i = tid; i < 16 * K; i += blockDim.x) {
        int r = i / K, c = i - r * K;
        int node = r0 + r;
        float v = (node < M) ? X[(size_t)node * K + c] : 0.f;
        sA[i] = (_Float16)v;
    }
    __syncthreads();
    const int ntiles = N >> 4, nk = K >> 5;
    v8f acc = {};
    for (int kt = 0; kt < nk; ++kt) {
        v16h a = load_a_frag(sA, lane, kt << 5, K);
        v16h b = *(const v16h*)(Wf + (((size_t)(kt * ntiles + wid)) << 9) + (lane << 4));
        acc = __builtin_amdgcn_wmma_f32_16x16x32_f16(false, a, false, b, (short)0, acc, false, false);
    }
    const int col = (wid << 4) + (lane & 15);
    const int hi8 = (lane >> 4) << 3;
    const float bv = bias[col];
#pragma unroll
    for (int r = 0; r < 8; ++r) {
        int node = r0 + r + hi8;
        if (node < M) {
            float v = acc[r] + bv;
            Y[(size_t)node * N + col] = v > 0.f ? v : 0.f;
        }
    }
}

// ---------------------------------------------------------------------------
// Fused SAGE update for one dst tile of 16 nodes (block = 256 thr = 8 waves):
//   lin = [agg/cnt , x_dst] @ [wl;wr] + bl     (K=256, N=128, WMMA)
//   out = relu(lin / max(||lin||_2, eps)) + x_dst
// ---------------------------------------------------------------------------
__global__ void sage_update(const float* __restrict__ agg, const float* __restrict__ cnt,
                            const float* __restrict__ xd, const _Float16* __restrict__ Wf,
                            const float* __restrict__ bias, float* __restrict__ out, int M) {
    __shared__ __attribute__((aligned(16))) _Float16 sA[16 * 256];
    __shared__ float sRes[16 * 128];
    __shared__ float sPart[8 * 16];
    __shared__ float sNorm[16];
    const int tile = blockIdx.x, tid = threadIdx.x;
    const int lane = tid & 31, wid = tid >> 5;
    const int r0 = tile << 4;

    for (int i = tid; i < 16 * 128; i += 256) {
        int r = i >> 7, c = i & 127;
        int node = r0 + r;
        float av = 0.f, xv = 0.f;
        if (node < M) {
            float inv = 1.f / fmaxf(cnt[node], 1.f);
            av = agg[(size_t)node * HDIM + c] * inv;
            xv = xd[(size_t)node * HDIM + c];
        }
        sA[r * 256 + c]       = (_Float16)av;
        sA[r * 256 + 128 + c] = (_Float16)xv;
        sRes[i] = xv;
    }
    __syncthreads();

    v8f acc = {};
#pragma unroll
    for (int kt = 0; kt < 8; ++kt) {
        v16h a = load_a_frag(sA, lane, kt << 5, 256);
        v16h b = *(const v16h*)(Wf + (((size_t)(kt * 8 + wid)) << 9) + (lane << 4));
        acc = __builtin_amdgcn_wmma_f32_16x16x32_f16(false, a, false, b, (short)0, acc, false, false);
    }

    const int nl  = lane & 15;
    const int col = (wid << 4) + nl;
    const int hi8 = (lane >> 4) << 3;
    const float bv = bias[col];
    float vv[8];
#pragma unroll
    for (int r = 0; r < 8; ++r) {
        float v = acc[r] + bv;
        vv[r] = v;
        float s = v * v;                       // reduce over 16 lanes sharing this row
#pragma unroll
        for (int m = 1; m < 16; m <<= 1) s += __shfl_xor(s, m, 32);
        if (nl == 0) sPart[wid * 16 + r + hi8] = s;
    }
    __syncthreads();
    if (tid < 16) {
        float t = 0.f;
#pragma unroll
        for (int w = 0; w < 8; ++w) t += sPart[w * 16 + tid];
        sNorm[tid] = 1.f / fmaxf(sqrtf(t), 1e-12f);
    }
    __syncthreads();
#pragma unroll
    for (int r = 0; r < 8; ++r) {
        int row = r + hi8;
        int node = r0 + row;
        if (node < M) {
            float v = vv[r] * sNorm[row];
            v = v > 0.f ? v : 0.f;
            out[(size_t)node * HDIM + col] = v + sRes[row * 128 + col];
        }
    }
}

// ---------------------------------------------------------------------------
// Head: z = relu(hu @ W1[128,64] + b1)  (WMMA);  out = z @ W2[64,8] + b2 (VALU)
// Block = 128 threads (4 waves), one 16-row tile.
// ---------------------------------------------------------------------------
__global__ void head_kernel(const float* __restrict__ hu, const _Float16* __restrict__ W1f,
                            const float* __restrict__ b1, const float* __restrict__ W2,
                            const float* __restrict__ b2, float* __restrict__ out, int M) {
    __shared__ __attribute__((aligned(16))) _Float16 sA[16 * 128];
    __shared__ float sZ[16 * 64];
    const int tile = blockIdx.x, tid = threadIdx.x;
    const int lane = tid & 31, wid = tid >> 5;
    const int r0 = tile << 4;
    for (int i = tid; i < 16 * 128; i += 128) {
        int r = i >> 7, c = i & 127;
        int node = r0 + r;
        sA[i] = (_Float16)((node < M) ? hu[(size_t)node * HDIM + c] : 0.f);
    }
    __syncthreads();
    v8f acc = {};
#pragma unroll
    for (int kt = 0; kt < 4; ++kt) {
        v16h a = load_a_frag(sA, lane, kt << 5, 128);
        v16h b = *(const v16h*)(W1f + (((size_t)(kt * 4 + wid)) << 9) + (lane << 4));
        acc = __builtin_amdgcn_wmma_f32_16x16x32_f16(false, a, false, b, (short)0, acc, false, false);
    }
    const int col = (wid << 4) + (lane & 15);
    const int hi8 = (lane >> 4) << 3;
    const float bv = b1[col];
#pragma unroll
    for (int r = 0; r < 8; ++r) {
        float v = acc[r] + bv;
        sZ[(r + hi8) * 64 + col] = v > 0.f ? v : 0.f;
    }
    __syncthreads();
    int row = tid >> 3, oc = tid & 7;
    int node = r0 + row;
    if (node < M) {
        float s = b2[oc];
#pragma unroll
        for (int k = 0; k < 64; ++k) s += sZ[row * 64 + k] * W2[k * 8 + oc];
        out[(size_t)node * 8 + oc] = s;
    }
}

// ---------------------------------------------------------------------------
extern "C" void kernel_launch(void* const* d_in, const int* in_sizes, int n_in,
                              void* d_out, int out_size, void* d_ws, size_t ws_size,
                              hipStream_t stream) {
    const float* x_user     = (const float*)d_in[0];
    const float* x_item     = (const float*)d_in[1];
    const int*   ei_u2i     = (const int*)d_in[2];
    const int*   ei_i2u     = (const int*)d_in[3];
    const float* enc_user_w = (const float*)d_in[4];
    const float* enc_user_b = (const float*)d_in[5];
    const float* enc_item_w = (const float*)d_in[6];
    const float* enc_item_b = (const float*)d_in[7];
    const float* wl[4] = {(const float*)d_in[8],  (const float*)d_in[11],
                          (const float*)d_in[14], (const float*)d_in[17]};
    const float* bl[4] = {(const float*)d_in[9],  (const float*)d_in[12],
                          (const float*)d_in[15], (const float*)d_in[18]};
    const float* wr[4] = {(const float*)d_in[10], (const float*)d_in[13],
                          (const float*)d_in[16], (const float*)d_in[19]};
    const float* head_w1 = (const float*)d_in[20];
    const float* head_b1 = (const float*)d_in[21];
    const float* head_w2 = (const float*)d_in[22];
    const float* head_b2 = (const float*)d_in[23];
    float* outp = (float*)d_out;

    // workspace carve-up (all fits in ~129 MB; entire hot set fits in 192MB L2)
    char* ws = (char*)d_ws;
    size_t off = 0;
    auto alloc = [&](size_t bytes) -> void* {
        void* p = ws + off;
        off = (off + bytes + 255) & ~(size_t)255;
        return p;
    };
    float* hu0 = (float*)alloc((size_t)NUSER * HDIM * 4);
    float* hu1 = (float*)alloc((size_t)NUSER * HDIM * 4);
    float* hi0 = (float*)alloc((size_t)NITEM * HDIM * 4);
    float* hi1 = (float*)alloc((size_t)NITEM * HDIM * 4);
    float* agg = (float*)alloc((size_t)NUSER * HDIM * 4);
    float* cnt = (float*)alloc((size_t)NUSER * 4);
    _Float16* wf_enc_u = (_Float16*)alloc((size_t)64 * 128 * 2);
    _Float16* wf_enc_i = (_Float16*)alloc((size_t)32 * 128 * 2);
    _Float16* wf_rel[4];
    for (int i = 0; i < 4; ++i) wf_rel[i] = (_Float16*)alloc((size_t)256 * 128 * 2);
    _Float16* wf_h1 = (_Float16*)alloc((size_t)128 * 64 * 2);

    // 1) pre-shuffle weights into WMMA B-fragment order
    auto shf = [&](const float* W, _Float16* dst, int K, int N) {
        int total = K * N;
        shuffle_weight<<<(total + 255) / 256, 256, 0, stream>>>(W, dst, K, N);
    };
    shf(enc_user_w, wf_enc_u, 64, 128);
    shf(enc_item_w, wf_enc_i, 32, 128);
    for (int i = 0; i < 4; ++i) {
        shf(wl[i], wf_rel[i], 128, 128);                 // K rows 0..127  (agg @ wl)
        shf(wr[i], wf_rel[i] + 128 * 128, 128, 128);     // K rows 128..255 (x_dst @ wr)
    }
    shf(head_w1, wf_h1, 128, 64);

    // 2) encoders
    gemm_bias_relu<<<(NUSER + 15) / 16, 256, 16 * 64 * 2, stream>>>(
        x_user, wf_enc_u, enc_user_b, hu0, NUSER, 64, 128);
    gemm_bias_relu<<<(NITEM + 15) / 16, 256, 16 * 32 * 2, stream>>>(
        x_item, wf_enc_i, enc_item_b, hi0, NITEM, 32, 128);

    // 3) two hetero-SAGE layers (ping-pong; both relations read OLD embeddings)
    float* hu_in = hu0; float* hi_in = hi0;
    float* hu_out = hu1; float* hi_out = hi1;
    const int scatter_blocks = (int)(((size_t)NEDGE * 32 + 255) / 256);
    for (int l = 0; l < 2; ++l) {
        int r = l * 2;
        // u2i: users -> items
        zero_f32<<<2048, 256, 0, stream>>>(agg, (size_t)NITEM * HDIM);
        zero_f32<<<64, 256, 0, stream>>>(cnt, (size_t)NITEM);
        scatter_sum<<<scatter_blocks, 256, 0, stream>>>(ei_u2i, ei_u2i + NEDGE, hu_in, agg, cnt, NEDGE);
        sage_update<<<(NITEM + 15) / 16, 256, 0, stream>>>(agg, cnt, hi_in, wf_rel[r], bl[r], hi_out, NITEM);
        // i2u: items -> users
        zero_f32<<<2048, 256, 0, stream>>>(agg, (size_t)NUSER * HDIM);
        zero_f32<<<64, 256, 0, stream>>>(cnt, (size_t)NUSER);
        scatter_sum<<<scatter_blocks, 256, 0, stream>>>(ei_i2u, ei_i2u + NEDGE, hi_in, agg, cnt, NEDGE);
        sage_update<<<(NUSER + 15) / 16, 256, 0, stream>>>(agg, cnt, hu_in, wf_rel[r + 1], bl[r + 1], hu_out, NUSER);
        // swap ping-pong
        float* t;
        t = hu_in; hu_in = hu_out; hu_out = t;
        t = hi_in; hi_in = hi_out; hi_out = t;
    }

    // 4) MLP head on users
    head_kernel<<<(NUSER + 15) / 16, 128, 0, stream>>>(
        hu_in, wf_h1, head_b1, head_w2, head_b2, outp, NUSER);
}
